// AttentionModel_39848706572724
// MI455X (gfx1250) — matrix operand order repaired and checked by reading, not compiled, not run on hardware
//
#include <hip/hip_runtime.h>
#include <hip/hip_bf16.h>

#define B_ 2
#define S_ 2048
#define H_ 16
#define D_ 64
#define LSTR 72  // padded LDS stride in halves (144B; matches TDM pad 4 dwords / 32 dwords)

typedef __attribute__((ext_vector_type(16))) _Float16 v16h;
typedef __attribute__((ext_vector_type(8)))  _Float16 v8h;
typedef __attribute__((ext_vector_type(4)))  _Float16 v4h;
typedef __attribute__((ext_vector_type(8)))  float    v8f;
typedef __attribute__((ext_vector_type(4)))  float    v4f;

typedef __attribute__((ext_vector_type(4))) unsigned int tdm_u4;
typedef __attribute__((ext_vector_type(8))) int          tdm_i8;
typedef __attribute__((ext_vector_type(4))) int          tdm_i4;

#if __has_builtin(__builtin_amdgcn_tensor_load_to_lds) && __has_builtin(__builtin_amdgcn_s_wait_tensorcnt)
#define USE_TDM 1
#else
#define USE_TDM 0
#endif

// A-matrix fragment (16x32 f16): lanes 0-15: rows M=0..15, K=k0+0..7 (v0-3), k0+16..23 (v4-7);
// lanes 16-31: K=k0+8..15 and k0+24..31. (ISA 7.12.2)
__device__ __forceinline__ v16h frag_ld_a(const _Float16* base, int row, int k0, int lane) {
  int kb = k0 + ((lane & 16) ? 8 : 0);
  const _Float16* p = base + row * LSTR + kb;
  v8h lo = *(const v8h*)(p);
  v8h hi = *(const v8h*)(p + 16);
  return __builtin_shufflevector(lo, hi, 0,1,2,3,4,5,6,7,8,9,10,11,12,13,14,15);
}

// B-matrix fragment (32x16 f16): lane = column; half-wave 0 holds K=k0+0..7,16..23, half-wave 1 K=k0+8..15,24..31.
__device__ __forceinline__ v16h frag_ld_b(const _Float16* base, int col, int k0, int lane) {
  const _Float16* p = base + col * LSTR + k0 + ((lane & 16) ? 8 : 0);
  v8h lo = *(const v8h*)(p);
  v8h hi = *(const v8h*)(p + 16);
  return __builtin_shufflevector(lo, hi, 0,1,2,3,4,5,6,7,8,9,10,11,12,13,14,15);
}

#if USE_TDM
__device__ __forceinline__ unsigned lds_byte_off(const void* p) {
  return (unsigned)(unsigned long long)
      (__attribute__((address_space(3))) const void*)(p);
}

// 2D f16 tile load: 64x64 elements, LDS rows padded 64->72 halves via TDM padding.
__device__ __forceinline__ void tdm_load_tile64(const _Float16* gptr, unsigned ldsOff,
                                                unsigned tensorW, unsigned tensorH,
                                                unsigned strideElems) {
  unsigned long long ga = (unsigned long long)gptr;
  tdm_u4 g0;
  g0[0] = 1u;                                            // count=1, user mode
  g0[1] = ldsOff;                                        // lds_addr
  g0[2] = (unsigned)(ga & 0xffffffffu);                  // global_addr lo
  g0[3] = (unsigned)((ga >> 32) & 0x1ffffffu) | (2u << 30); // global_addr hi | type=2
  tdm_i8 g1;
  // data_size=2B (code 1), pad_enable, pad_interval=4 (32 dwords), pad_amount=3 (4 dwords)
  g1[0] = (int)((1u << 16) | (1u << 20) | (4u << 22) | (3u << 25));
  g1[1] = (int)(tensorW << 16);                          // tensor_dim0[15:0]
  g1[2] = (int)((tensorW >> 16) | (tensorH << 16));      // tensor_dim0[31:16] | tensor_dim1[15:0]
  g1[3] = (int)((tensorH >> 16) | (64u << 16));          // tensor_dim1[31:16] | tile_dim0=64
  g1[4] = (int)64u;                                      // tile_dim1=64, tile_dim2=0
  g1[5] = (int)strideElems;                              // tensor_dim0_stride lo32
  g1[6] = 0;
  g1[7] = 0;
  tdm_i4 z4 = {0, 0, 0, 0};
#if __clang_major__ >= 23
  tdm_i8 z8 = {0, 0, 0, 0, 0, 0, 0, 0};
  __builtin_amdgcn_tensor_load_to_lds(g0, g1, z4, z4, z8, 0);
#else
  __builtin_amdgcn_tensor_load_to_lds(g0, g1, z4, z4, 0);
#endif
}
#endif

// ---------------- per-head linear projection:  out = x @ W^T + b ----------------
__global__ __launch_bounds__(128)
void proj_kernel(const float* __restrict__ x, const float* __restrict__ W,
                 const float* __restrict__ bias, _Float16* __restrict__ out,
                 int transposeOut) {
  __shared__ __align__(16) _Float16 xs[64 * LSTR];
  __shared__ __align__(16) _Float16 ws[64 * LSTR];
  const int tid  = threadIdx.x;
  const int lane = tid & 31;
  const int wave = tid >> 5;
  const long m0  = (long)blockIdx.x * 64;

#pragma unroll
  for (int i = 0; i < 8; ++i) {
    int c    = i * 128 + tid;
    int row  = c >> 4;
    int col4 = (c & 15) * 4;
    v4f xv = *(const v4f*)(x + (m0 + row) * 64 + col4);
    v4f wv = *(const v4f*)(W + row * 64 + col4);
    v4h xh = { (_Float16)xv[0], (_Float16)xv[1], (_Float16)xv[2], (_Float16)xv[3] };
    v4h wh = { (_Float16)wv[0], (_Float16)wv[1], (_Float16)wv[2], (_Float16)wv[3] };
    *(v4h*)(&xs[row * LSTR + col4]) = xh;
    *(v4h*)(&ws[row * LSTR + col4]) = wh;
  }
  __syncthreads();

  const int rowA  = wave * 16 + (lane & 15);
  const int hiRow = (lane >> 4) * 8;
  v16h a0 = frag_ld_a(xs, rowA, 0, lane);
  v16h a1 = frag_ld_a(xs, rowA, 32, lane);

#pragma unroll
  for (int t = 0; t < 4; ++t) {
    int col = t * 16 + (lane & 15);
    v16h b0 = frag_ld_b(ws, col, 0, lane);
    v16h b1 = frag_ld_b(ws, col, 32, lane);
    v8f acc = {};
    acc = __builtin_amdgcn_wmma_f32_16x16x32_f16(false, a0, false, b0, (short)0, acc, false, false);
    acc = __builtin_amdgcn_wmma_f32_16x16x32_f16(false, a1, false, b1, (short)0, acc, false, false);
    float bv = bias[col];
#pragma unroll
    for (int r = 0; r < 8; ++r) {
      long m = m0 + wave * 16 + r + hiRow;
      int h = (int)(m & (H_ - 1));
      int s = (int)((m >> 4) & (S_ - 1));
      int b = (int)(m >> 15);
      float v = acc[r] + bv;
      long off = transposeOut
          ? ((long)(b * H_ + h) * D_ + col) * S_ + s
          : ((long)(b * H_ + h) * S_ + s) * D_ + col;
      out[off] = (_Float16)v;
    }
  }
}

// ---------------- flash attention, TDM double-buffered K/V tiles ----------------
__global__ __launch_bounds__(128)
void attn_kernel(const _Float16* __restrict__ Qh, const _Float16* __restrict__ Kh,
                 const _Float16* __restrict__ Vt, const float* __restrict__ mask,
                 float* __restrict__ out) {
  __shared__ __align__(16) _Float16 kbuf[2][64 * LSTR];
  __shared__ __align__(16) _Float16 vbuf[2][64 * LSTR];
  __shared__ __align__(16) _Float16 pbuf[64 * LSTR];

  const int tid  = threadIdx.x;
  const int lane = tid & 31;
  const int wave = tid >> 5;
  const int qt = blockIdx.x & 31;
  const int bh = blockIdx.x >> 5;
  const int b  = bh >> 4, h = bh & 15;
  const int q0 = qt * 64;
  const int laneCol = lane & 15;
  const int hiRow   = (lane >> 4) * 8;

  const _Float16* Qbase = Qh + ((long)bh * S_ + q0) * D_;
  const _Float16* Kbase = Kh + (long)bh * S_ * D_;
  const _Float16* Vbase = Vt + (long)bh * D_ * S_;

#if USE_TDM
  // issue tile 0 via the Tensor Data Mover as early as possible (wave 0 only)
  if (wave == 0) {
    tdm_load_tile64(Kbase, lds_byte_off(&kbuf[0][0]), D_, S_, D_);          // K rows: stride D
    tdm_load_tile64(Vbase, lds_byte_off(&vbuf[0][0]), S_, D_, S_);          // V^T rows: stride S
  }
#endif

  // stage the 64x64 Q tile, pull the wave's two A fragments into registers
#pragma unroll
  for (int i = 0; i < 4; ++i) {
    int c = i * 128 + tid;
    int row = c >> 3, col8 = (c & 7) * 8;
    *(v8h*)(&pbuf[row * LSTR + col8]) = *(const v8h*)(Qbase + row * D_ + col8);
  }
  __syncthreads();
  const int rA = wave * 16 + laneCol;
  v16h aQ0 = frag_ld_a(pbuf, rA, 0, lane);
  v16h aQ1 = frag_ld_a(pbuf, rA, 32, lane);

  v8f O[4];
#pragma unroll
  for (int t = 0; t < 4; ++t) { v8f z = {}; O[t] = z; }
  float mrow[8], lrow[8];
  const float* mptr[8];
  unsigned zrow[8];
#pragma unroll
  for (int r = 0; r < 8; ++r) {
    mrow[r] = -1e30f; lrow[r] = 0.f;
    int q = q0 + wave * 16 + r + hiRow;
    mptr[r] = mask + ((long)b * S_ + q) * S_;                       // loop-invariant row ptr
    zrow[r] = (unsigned)((b * H_ + h) * S_ + q) * (unsigned)S_;     // dropout hash base
  }

  for (int kt = 0; kt < 32; ++kt) {
    const int s0 = kt * 64;
    const int cur = kt & 1;
    const _Float16* kb_ = kbuf[cur];
    const _Float16* vb_ = vbuf[cur];

#if USE_TDM
    if (wave == 0) __builtin_amdgcn_s_wait_tensorcnt(0);   // current tile landed in LDS
    __syncthreads();                                       // publish to all waves
    if (wave == 0 && kt + 1 < 32) {                        // overlap next tile with compute
      int nxt = (kt + 1) & 1;
      tdm_load_tile64(Kbase + (long)(s0 + 64) * D_, lds_byte_off(&kbuf[nxt][0]), D_, S_, D_);
      tdm_load_tile64(Vbase + (s0 + 64),            lds_byte_off(&vbuf[nxt][0]), S_, D_, S_);
    }
#else
    __syncthreads();                                       // previous tile fully consumed
#pragma unroll
    for (int i = 0; i < 4; ++i) {
      int c = i * 128 + tid;
      int row = c >> 3, col8 = (c & 7) * 8;
      *(v8h*)(&kbuf[cur][row * LSTR + col8]) = *(const v8h*)(Kbase + (long)(s0 + row) * D_ + col8);
      *(v8h*)(&vbuf[cur][row * LSTR + col8]) = *(const v8h*)(Vbase + (long)row * S_ + s0 + col8);
    }
    if (kt + 1 < 32)
      __builtin_prefetch(Kbase + (long)(s0 + 64) * D_ + tid * 32, 0, 1);
    __syncthreads();
#endif

    // S = Q K^T / sqrt(D) + mask
    v8f sc[4];
#pragma unroll
    for (int t = 0; t < 4; ++t) {
      int col = t * 16 + laneCol;
      v16h b0 = frag_ld_b(kb_, col, 0, lane);
      v16h b1 = frag_ld_b(kb_, col, 32, lane);
      v8f acc = {};
      acc = __builtin_amdgcn_wmma_f32_16x16x32_f16(false, aQ0, false, b0, (short)0, acc, false, false);
      acc = __builtin_amdgcn_wmma_f32_16x16x32_f16(false, aQ1, false, b1, (short)0, acc, false, false);
      int kcol = s0 + col;
#pragma unroll
      for (int r = 0; r < 8; ++r)
        acc[r] = acc[r] * 0.125f + mptr[r][kcol];
      sc[t] = acc;
    }

    // online softmax (rows live in 16-lane half-groups)
#pragma unroll
    for (int r = 0; r < 8; ++r) {
      float vmax = fmaxf(fmaxf(sc[0][r], sc[1][r]), fmaxf(sc[2][r], sc[3][r]));
      vmax = fmaxf(vmax, __shfl_xor(vmax, 1));
      vmax = fmaxf(vmax, __shfl_xor(vmax, 2));
      vmax = fmaxf(vmax, __shfl_xor(vmax, 4));
      vmax = fmaxf(vmax, __shfl_xor(vmax, 8));
      float mnew = fmaxf(mrow[r], vmax);
      float corr = __expf(mrow[r] - mnew);
      mrow[r] = mnew;
      float rsum = 0.f;
#pragma unroll
      for (int t = 0; t < 4; ++t) {
        float p = __expf(sc[t][r] - mnew);
        sc[t][r] = p;
        rsum += p;
      }
      rsum += __shfl_xor(rsum, 1);
      rsum += __shfl_xor(rsum, 2);
      rsum += __shfl_xor(rsum, 4);
      rsum += __shfl_xor(rsum, 8);
      lrow[r] = lrow[r] * corr + rsum;     // denominator: pre-dropout probabilities
#pragma unroll
      for (int t = 0; t < 4; ++t) O[t][r] *= corr;
    }

    // dropout (deterministic hash) + C/D -> A re-layout via per-wave LDS scratch
#pragma unroll
    for (int t = 0; t < 4; ++t) {
      int kcol = s0 + t * 16 + laneCol;
#pragma unroll
      for (int r = 0; r < 8; ++r) {
        unsigned z = zrow[r] + (unsigned)kcol;
        z ^= z >> 16; z *= 0x7feb352dU; z ^= z >> 15; z *= 0x846ca68bU; z ^= z >> 16;
        float p = sc[t][r];
        p = ((z & 0xFFFFFFu) >= 1677722u) ? p * (1.0f / 0.9f) : 0.0f;
        pbuf[(wave * 16 + r + hiRow) * LSTR + t * 16 + laneCol] = (_Float16)p;
      }
    }
    __asm__ volatile("s_wait_dscnt 0" ::: "memory");  // same-wave DS RAW fence

    // O += P V
    v16h aP0 = frag_ld_a(pbuf, rA, 0, lane);
    v16h aP1 = frag_ld_a(pbuf, rA, 32, lane);
#pragma unroll
    for (int t = 0; t < 4; ++t) {
      int col = t * 16 + laneCol;
      v16h b0 = frag_ld_b(vb_, col, 0, lane);
      v16h b1 = frag_ld_b(vb_, col, 32, lane);
      O[t] = __builtin_amdgcn_wmma_f32_16x16x32_f16(false, aP0, false, b0, (short)0, O[t], false, false);
      O[t] = __builtin_amdgcn_wmma_f32_16x16x32_f16(false, aP1, false, b1, (short)0, O[t], false, false);
    }
  }

  // normalize and store [B,H,S,D] f32
#pragma unroll
  for (int t = 0; t < 4; ++t) {
    int d = t * 16 + laneCol;
#pragma unroll
    for (int r = 0; r < 8; ++r) {
      int q = q0 + wave * 16 + r + hiRow;
      out[((long)bh * S_ + q) * D_ + d] = O[t][r] / lrow[r];
    }
  }
}

extern "C" void kernel_launch(void* const* d_in, const int* in_sizes, int n_in,
                              void* d_out, int out_size, void* d_ws, size_t ws_size,
                              hipStream_t stream) {
  (void)in_sizes; (void)n_in; (void)out_size;
  const float* query = (const float*)d_in[0];
  const float* key_  = (const float*)d_in[1];
  const float* value = (const float*)d_in[2];
  const float* mask  = (const float*)d_in[3];
  const float* Wq = (const float*)d_in[4];
  const float* bq = (const float*)d_in[5];
  const float* Wk = (const float*)d_in[6];
  const float* bk = (const float*)d_in[7];
  const float* Wv = (const float*)d_in[8];
  const float* bv = (const float*)d_in[9];
  float* out = (float*)d_out;

  const size_t n = (size_t)B_ * H_ * S_ * D_;
  if (ws_size < 3 * n * sizeof(_Float16)) return;
  _Float16* Qh = (_Float16*)d_ws;
  _Float16* Kh = Qh + n;
  _Float16* Vt = Kh + n;

  const int mblocks = (B_ * S_ * H_) / 64;      // 1024
  const int ablocks = B_ * H_ * (S_ / 64);      // 1024
  dim3 blk(128);
  proj_kernel<<<mblocks, blk, 0, stream>>>(query, Wq, bq, Qh, 0);
  proj_kernel<<<mblocks, blk, 0, stream>>>(key_,  Wk, bk, Kh, 0);
  proj_kernel<<<mblocks, blk, 0, stream>>>(value, Wv, bv, Vt, 1);
  attn_kernel<<<ablocks, blk, 0, stream>>>(Qh, Kh, Vt, mask, out);
}
